// GraphPreModel_81913616270030
// MI455X (gfx1250) — matrix operand order repaired and checked by reading, not compiled, not run on hardware
//
#include <hip/hip_runtime.h>

// ---------------------------------------------------------------------------
// GraphPreModel for MI455X (gfx1250): bf16 WMMA GEMMs + flash attention.
// Precision: bf16 inputs, f32 WMMA accumulation, f32 residual/LayerNorm.
// Workspace use: ~64.5 MB (weights-in-bf16 + activations, embed temps aliased).
// gfx1250 features: v_wmma_f32_16x16x32_bf16, global_load_async_to_lds_b128
// (ASYNCcnt) for weight-tile staging, global_prefetch, split wait counters.
// ---------------------------------------------------------------------------

#define TRY_ASYNC_LDS 1

typedef __attribute__((ext_vector_type(16))) __bf16   v16bf_t;
typedef __attribute__((ext_vector_type(8)))  float    v8f_t;
typedef __attribute__((ext_vector_type(8)))  unsigned v8u_t;
typedef unsigned short bh16;   // raw bf16 bits

#define DEV static __device__ __forceinline__

DEV bh16 f2bf(float f) {
  unsigned u = __builtin_bit_cast(unsigned, f);
  u += 0x7FFFu + ((u >> 16) & 1u);           // round-to-nearest-even
  return (bh16)(u >> 16);
}
DEV unsigned packbf(float a, float b) {      // two f32 -> packed bf16x2
  return (unsigned)f2bf(a) | ((unsigned)f2bf(b) << 16);
}

DEV v8f_t wmma_bf16(v8u_t a, v8u_t b, v8f_t c) {
  return __builtin_amdgcn_wmma_f32_16x16x32_bf16(
      /*neg_a=*/false, __builtin_bit_cast(v16bf_t, a),
      /*neg_b=*/false, __builtin_bit_cast(v16bf_t, b),
      /*c_mod=*/(short)0, c, /*reuse_a=*/false, /*reuse_b=*/false);
}

DEV float halfmax(float x) {            // reduce over lanes 0-15 / 16-31 halves
#pragma unroll
  for (int o = 1; o < 16; o <<= 1) x = fmaxf(x, __shfl_xor(x, o, 32));
  return x;
}
DEV float halfsum(float x) {
#pragma unroll
  for (int o = 1; o < 16; o <<= 1) x += __shfl_xor(x, o, 32);
  return x;
}

// ---------------------------------------------------------------------------
// f32 -> bf16 bulk conversion (weights)
// ---------------------------------------------------------------------------
__global__ void conv_k(const float* __restrict__ s, bh16* __restrict__ d, int n) {
  int i = blockIdx.x * 256 + threadIdx.x;
  if (i < n) d[i] = f2bf(s[i]);
}

// opcode embedding gather -> first 256 cols of hcat (ld 512)
__global__ void gather_k(const int* __restrict__ opcode, const float* __restrict__ Wop,
                         bh16* __restrict__ hcat) {
  int i = blockIdx.x * 256 + threadIdx.x;           // over 8184*256
  if (i < 8184 * 256) {
    int m = i >> 8, c = i & 255;
    hcat[(long)m * 512 + c] = f2bf(Wop[(long)opcode[m] * 256 + c]);
  }
}

// build x (B,1024,256) f32 + bf16 mirror, inserting CLS token at row 0
__global__ void assemble_k(const float* __restrict__ hbuf, const float* __restrict__ cls,
                           float* __restrict__ x, bh16* __restrict__ xb) {
  int i = blockIdx.x * 256 + threadIdx.x;           // over 8192*256
  int row = i >> 8, c = i & 255;
  int b = row >> 10, s = row & 1023;
  float v = (s == 0) ? cls[c] : hbuf[((long)(b * 1023 + s - 1)) * 256 + c];
  x[i] = v;
  xb[i] = f2bf(v);
}

// ---------------------------------------------------------------------------
// Tiled bf16 WMMA GEMM: C = epilogue(A @ W + bias [+res])
//   A: MxK (f32 or bf16 bits), W: KxN bf16 row-major, outputs f32 and/or bf16
//   Block tile 128x128x32, 8 waves (2 M x 4 N), 8 v_wmma per wave per K-step.
//   A staged with b128 loads/stores; W tile staged via async-to-LDS b128.
// ---------------------------------------------------------------------------
template <bool A_F32, bool DO_GELU, bool HAS_RES>
__global__ __launch_bounds__(256) void gemm_k(
    const void* __restrict__ Ap, const bh16* __restrict__ W,
    const float* __restrict__ bias, const float* __restrict__ res,
    float* __restrict__ Cf, bh16* __restrict__ Cb, int M, int K, int N, int ldc) {
  __shared__ bh16 As[128][48];   // [row][k], stride 96B (16B aligned segs)
  __shared__ bh16 Bs[32][136];   // [k][col], stride 272B (16B aligned segs)

  const int tid  = threadIdx.x;
  const int lane = tid & 31, wid = tid >> 5;
  const int half = lane >> 4, l15 = lane & 15;
  const int mBase = blockIdx.y * 128;
  const int nBase = blockIdx.x * 128;
  const int waveM = (wid & 1) * 64;
  const int waveN = (wid >> 1) * 32;

  v8f_t acc[4][2];
  const v8f_t zf = {0.f, 0.f, 0.f, 0.f, 0.f, 0.f, 0.f, 0.f};
#pragma unroll
  for (int mi = 0; mi < 4; ++mi)
#pragma unroll
    for (int ni = 0; ni < 2; ++ni) acc[mi][ni] = zf;

  const int arow  = tid >> 1;        // 0..127
  const int akseg = (tid & 1) * 16;
  const int brow  = tid >> 3;        // 0..31 (k row)
  const int bcseg = (tid & 7) * 16;
  const bool aRowOK = (mBase + arow) < M;
  const long aRowG  = (long)(mBase + arow);

  const int kTiles = (K + 31) >> 5;
  for (int kt = 0; kt < kTiles; ++kt) {
    const int k0 = kt << 5;
    // ---- stage A tile (b128 fast path; guarded path on K tail / M edge) ----
    if (aRowOK && (k0 + 32 <= K)) {
      if (A_F32) {
        const float4* ap = (const float4*)((const float*)Ap + aRowG * K + k0 + akseg);
        float4 f0 = ap[0], f1 = ap[1], f2 = ap[2], f3 = ap[3];
        uint4 u0, u1;
        u0.x = packbf(f0.x, f0.y); u0.y = packbf(f0.z, f0.w);
        u0.z = packbf(f1.x, f1.y); u0.w = packbf(f1.z, f1.w);
        u1.x = packbf(f2.x, f2.y); u1.y = packbf(f2.z, f2.w);
        u1.z = packbf(f3.x, f3.y); u1.w = packbf(f3.z, f3.w);
        *(uint4*)&As[arow][akseg]     = u0;
        *(uint4*)&As[arow][akseg + 8] = u1;
      } else {
        const uint4* ap = (const uint4*)((const bh16*)Ap + aRowG * K + k0 + akseg);
        *(uint4*)&As[arow][akseg]     = ap[0];
        *(uint4*)&As[arow][akseg + 8] = ap[1];
      }
    } else {
#pragma unroll
      for (int j = 0; j < 16; ++j) {
        int kk = k0 + akseg + j;
        bh16 vb = 0;
        if (aRowOK && kk < K) {
          if (A_F32) vb = f2bf(((const float*)Ap)[aRowG * K + kk]);
          else       vb = ((const bh16*)Ap)[aRowG * K + kk];
        }
        As[arow][akseg + j] = vb;
      }
    }
    // ---- stage W tile: async copy global -> LDS (ASYNCcnt path) ----
    if (k0 + brow < K) {
      const bh16* gp = W + (long)(k0 + brow) * N + nBase + bcseg;
#if TRY_ASYNC_LDS
      unsigned lds0 = (unsigned)(size_t)&Bs[brow][bcseg];
      asm volatile(
          "global_load_async_to_lds_b128 %0, %1, off\n\t"
          "global_load_async_to_lds_b128 %2, %3, off"
          :: "v"(lds0), "v"(gp), "v"(lds0 + 16u), "v"(gp + 8)
          : "memory");
#else
      *(uint4*)&Bs[brow][bcseg]     = ((const uint4*)gp)[0];
      *(uint4*)&Bs[brow][bcseg + 8] = ((const uint4*)gp)[1];
#endif
    } else {
      const uint4 z4 = {0u, 0u, 0u, 0u};
      *(uint4*)&Bs[brow][bcseg]     = z4;
      *(uint4*)&Bs[brow][bcseg + 8] = z4;
    }
    // prefetch next A K-tile (global_prefetch path)
    if (kt + 1 < kTiles && aRowOK && (k0 + 32 + akseg) < K) {
      if (A_F32) __builtin_prefetch(&((const float*)Ap)[aRowG * K + k0 + 32 + akseg], 0, 1);
      else       __builtin_prefetch(&((const bh16*)Ap)[aRowG * K + k0 + 32 + akseg], 0, 1);
    }
#if TRY_ASYNC_LDS
    asm volatile("s_wait_asynccnt 0" ::: "memory");
#endif
    __syncthreads();
    // ---- compute: fragments from LDS, 8 WMMAs per wave ----
    v8u_t bfr[2];
#pragma unroll
    for (int ni = 0; ni < 2; ++ni) {
      const int col = waveN + ni * 16 + l15;
      const int kb2 = half * 16;               // B: lanes 0-15 K=0..15, 16-31 K=16..31
#pragma unroll
      for (int j = 0; j < 8; ++j) {
        unsigned lo = Bs[kb2 + 2 * j][col];
        unsigned hi = Bs[kb2 + 2 * j + 1][col];
        bfr[ni][j] = lo | (hi << 16);
      }
    }
#pragma unroll
    for (int mi = 0; mi < 4; ++mi) {
      const int row = waveM + mi * 16 + l15;
      v8u_t afr;
#pragma unroll
      for (int j = 0; j < 8; ++j) {            // A: lanes 0-15 K=0..15, 16-31 K=16..31
        int kk = half * 16 + 2 * j;
        afr[j] = *(const unsigned*)&As[row][kk];
      }
#pragma unroll
      for (int ni = 0; ni < 2; ++ni) acc[mi][ni] = wmma_bf16(afr, bfr[ni], acc[mi][ni]);
    }
    __syncthreads();
  }

  // ---- epilogue ----
#pragma unroll
  for (int mi = 0; mi < 4; ++mi) {
#pragma unroll
    for (int ni = 0; ni < 2; ++ni) {
      const int col = nBase + waveN + ni * 16 + l15;
#pragma unroll
      for (int r = 0; r < 8; ++r) {
        const int row = mBase + waveM + mi * 16 + half * 8 + r;
        if (row < M) {
          float v = acc[mi][ni][r] + bias[col];
          if (DO_GELU) v = 0.5f * v * (1.0f + erff(v * 0.70710678118654752f));
          if (HAS_RES) v += res[(long)row * ldc + col];
          const long idx = (long)row * ldc + col;
          if (Cf) Cf[idx] = v;
          if (Cb) Cb[idx] = f2bf(v);
        }
      }
    }
  }
}

// ---------------------------------------------------------------------------
// Flash attention: one wave handles a 16-query tile for one (b,h).
// 64 keys per iteration: 4 score WMMAs + online softmax + 4 P.V WMMAs.
// Bias looked up on the fly from dist_matrix + 32-entry table.
// ---------------------------------------------------------------------------
__global__ __launch_bounds__(32) void attn_k(
    const bh16* __restrict__ q, const bh16* __restrict__ k, const bh16* __restrict__ v,
    const int* __restrict__ dist, const float* __restrict__ table, bh16* __restrict__ ctx) {
  __shared__ float tab[32];
  __shared__ bh16  Pt[16][72];                  // 16 q x 64 keys, padded

  const int lane = threadIdx.x;
  const int half = lane >> 4, l15 = lane & 15;
  const int qt = blockIdx.x;                    // 0..63 (16-query tiles)
  const int bh = blockIdx.y;                    // 0..63
  const int b = bh >> 3, h = bh & 7;
  const int S = 1024, E = 256;
  const float scale = 0.17677669529663687f;     // 1/sqrt(32)

  tab[lane] = table[lane];
  __syncthreads();

  const int qBase = qt * 16;
  // Q A-fragment (16 x 32), loaded once
  const unsigned* qp = (const unsigned*)(q + ((long)(b * S + qBase + l15)) * E + h * 32);
  v8u_t qf;
#pragma unroll
  for (int j = 0; j < 8; ++j) {
    int kk = half * 16 + 2 * j;
    qf[j] = qp[kk >> 1];
  }

  const v8f_t zf = {0.f, 0.f, 0.f, 0.f, 0.f, 0.f, 0.f, 0.f};
  v8f_t c0 = zf, c1 = zf;                       // ctx accum, dims 0-15 / 16-31
  float m[8], l[8];
#pragma unroll
  for (int r = 0; r < 8; ++r) { m[r] = -1e30f; l[r] = 0.f; }

  for (int kb = 0; kb < S; kb += 64) {
    // ---- scores: 4 x (16q x 16k) WMMA tiles ----
    v8f_t s[4];
#pragma unroll
    for (int t = 0; t < 4; ++t) {
      const unsigned* kp =
          (const unsigned*)(k + ((long)(b * S + kb + t * 16 + l15)) * E + h * 32);
      v8u_t kf;
#pragma unroll
      for (int j = 0; j < 8; ++j) kf[j] = kp[half * 8 + j];
      s[t] = wmma_bf16(qf, kf, zf);
    }
    // ---- bias + scale + online softmax (per row r) ----
#pragma unroll
    for (int r = 0; r < 8; ++r) {
      const int qrow = qBase + half * 8 + r;
      const int* dp = dist + ((long)(b * S + qrow)) * S + kb;
      float sv[4];
#pragma unroll
      for (int t = 0; t < 4; ++t) {
        int d = dp[t * 16 + l15];
        int idx = ((unsigned)d > 30u) ? 31 : d;   // d==-1 or d>30 -> clamp to 31
        sv[t] = s[t][r] * scale + tab[idx];
      }
      float cm = halfmax(fmaxf(fmaxf(sv[0], sv[1]), fmaxf(sv[2], sv[3])));
      float mn = fmaxf(m[r], cm);
      float corr = __expf(m[r] - mn);
      m[r] = mn;
      float ps = 0.f;
#pragma unroll
      for (int t = 0; t < 4; ++t) {
        float p = __expf(sv[t] - mn);
        ps += p;
        Pt[half * 8 + r][t * 16 + l15] = f2bf(p);
      }
      l[r] = l[r] * corr + halfsum(ps);
      c0[r] *= corr; c1[r] *= corr;
    }
    __syncthreads();
    // ---- P.V: 2 key-chunks x 2 dim-tiles = 4 WMMAs ----
#pragma unroll
    for (int c = 0; c < 2; ++c) {
      v8u_t pf;
#pragma unroll
      for (int j = 0; j < 8; ++j) {
        int kk = c * 32 + half * 16 + 2 * j;
        pf[j] = *(const unsigned*)&Pt[l15][kk];
      }
      v8u_t vf0, vf1;
#pragma unroll
      for (int j = 0; j < 8; ++j) {
        const long r0 = ((long)(b * S + kb + c * 32 + half * 16 + 2 * j)) * E + h * 32;
        unsigned lo0 = v[r0 + l15],      hi0 = v[r0 + E + l15];
        unsigned lo1 = v[r0 + 16 + l15], hi1 = v[r0 + E + 16 + l15];
        vf0[j] = lo0 | (hi0 << 16);
        vf1[j] = lo1 | (hi1 << 16);
      }
      c0 = wmma_bf16(pf, vf0, c0);
      c1 = wmma_bf16(pf, vf1, c1);
    }
    __syncthreads();
  }

  // normalize + store ctx (token-major, head-interleaved -> feeds Wo GEMM)
#pragma unroll
  for (int r = 0; r < 8; ++r) {
    const int qrow = qBase + half * 8 + r;
    const float inv = 1.0f / l[r];
    const long o = ((long)(b * S + qrow)) * E + h * 32;
    ctx[o + l15]      = f2bf(c0[r] * inv);
    ctx[o + 16 + l15] = f2bf(c1[r] * inv);
  }
}

// ---------------------------------------------------------------------------
// LayerNorm over 256-dim rows: one wave per row (32 lanes x 8 values)
// ---------------------------------------------------------------------------
__global__ __launch_bounds__(256) void ln_k(
    const float* __restrict__ y, const float* __restrict__ g, const float* __restrict__ bta,
    float* __restrict__ xo, bh16* __restrict__ xbo) {
  const int lane = threadIdx.x & 31;
  const int row  = blockIdx.x * 8 + (threadIdx.x >> 5);
  const float* yr = y + (long)row * 256;
  float vals[8], s = 0.f;
#pragma unroll
  for (int j = 0; j < 8; ++j) { vals[j] = yr[j * 32 + lane]; s += vals[j]; }
#pragma unroll
  for (int o = 1; o < 32; o <<= 1) s += __shfl_xor(s, o, 32);
  const float mean = s * (1.0f / 256.0f);
  float vs = 0.f;
#pragma unroll
  for (int j = 0; j < 8; ++j) { float d = vals[j] - mean; vs += d * d; }
#pragma unroll
  for (int o = 1; o < 32; o <<= 1) vs += __shfl_xor(vs, o, 32);
  const float rs = rsqrtf(vs * (1.0f / 256.0f) + 1e-12f);
#pragma unroll
  for (int j = 0; j < 8; ++j) {
    int c = j * 32 + lane;
    float o = (vals[j] - mean) * rs * g[c] + bta[c];
    xo[(long)row * 256 + c]  = o;
    xbo[(long)row * 256 + c] = f2bf(o);
  }
}

// ---------------------------------------------------------------------------
extern "C" void kernel_launch(void* const* d_in, const int* in_sizes, int n_in,
                              void* d_out, int out_size, void* d_ws, size_t ws_size,
                              hipStream_t stream) {
  (void)in_sizes; (void)n_in; (void)out_size; (void)ws_size;
  const float* node_feat = (const float*)d_in[0];
  const int*   opcode    = (const int*)d_in[1];
  const int*   dist      = (const int*)d_in[2];
  const float* W_op      = (const float*)d_in[3];
  const float* W_feat    = (const float*)d_in[4];
  const float* b_feat    = (const float*)d_in[5];
  const float* W_mlp1    = (const float*)d_in[6];
  const float* b_mlp1    = (const float*)d_in[7];
  const float* W_mlp2    = (const float*)d_in[8];
  const float* b_mlp2    = (const float*)d_in[9];
  const float* cls       = (const float*)d_in[10];
  const float* dtab      = (const float*)d_in[11];
  const float* Wq = (const float*)d_in[12]; const float* bq = (const float*)d_in[13];
  const float* Wk = (const float*)d_in[14]; const float* bk = (const float*)d_in[15];
  const float* Wv = (const float*)d_in[16]; const float* bv = (const float*)d_in[17];
  const float* Wo = (const float*)d_in[18]; const float* bo = (const float*)d_in[19];
  const float* ln1g = (const float*)d_in[20]; const float* ln1b = (const float*)d_in[21];
  const float* Wf1 = (const float*)d_in[22]; const float* bf1 = (const float*)d_in[23];
  const float* Wf2 = (const float*)d_in[24]; const float* bf2 = (const float*)d_in[25];
  const float* ln2g = (const float*)d_in[26]; const float* ln2b = (const float*)d_in[27];

  const int Mn = 8184;    // B*N tokens (no CLS)
  const int Mt = 8192;    // B*S tokens

  // ---- workspace carve-up (~64.5 MB) ----
  char* w = (char*)d_ws; size_t off = 0;
  auto take = [&](size_t bytes) -> char* {
    char* p = w + off; off = (off + bytes + 255) & ~(size_t)255; return p;
  };
  bh16* wfeat = (bh16*)take(140 * 256 * 2);
  bh16* wmlp1 = (bh16*)take(512 * 256 * 2);
  bh16* wmlp2 = (bh16*)take(256 * 256 * 2);
  bh16* wq = (bh16*)take(6 * 256 * 256 * 2);
  bh16* wk = (bh16*)take(6 * 256 * 256 * 2);
  bh16* wv = (bh16*)take(6 * 256 * 256 * 2);
  bh16* wo = (bh16*)take(6 * 256 * 256 * 2);
  bh16* wf1w = (bh16*)take(6 * 256 * 1024 * 2);
  bh16* wf2w = (bh16*)take(6 * 1024 * 256 * 2);
  float* x  = (float*)take((size_t)Mt * 256 * 4);
  bh16*  xb = (bh16*)take((size_t)Mt * 256 * 2);
  bh16*  qb = (bh16*)take((size_t)Mt * 256 * 2);
  bh16*  kb = (bh16*)take((size_t)Mt * 256 * 2);
  bh16*  vb = (bh16*)take((size_t)Mt * 256 * 2);
  bh16*  ctxb = (bh16*)take((size_t)Mt * 256 * 2);
  float* y  = (float*)take((size_t)Mt * 256 * 4);
  bh16*  ffn = (bh16*)take((size_t)Mt * 1024 * 2);
  // embed-phase temps alias layer-phase buffers (disjoint in time):
  bh16*  hcat = qb;                // 8184*512 bf16  <= q+k region
  bh16*  t1   = vb;                // 8184*256 bf16  <= v region
  float* hbuf = y;                 // 8184*256 f32   <= y region

  auto conv = [&](const float* s, bh16* d, int n) {
    conv_k<<<(n + 255) / 256, 256, 0, stream>>>(s, d, n);
  };
  conv(W_feat, wfeat, 140 * 256);
  conv(W_mlp1, wmlp1, 512 * 256);
  conv(W_mlp2, wmlp2, 256 * 256);
  conv(Wq, wq, 6 * 256 * 256);
  conv(Wk, wk, 6 * 256 * 256);
  conv(Wv, wv, 6 * 256 * 256);
  conv(Wo, wo, 6 * 256 * 256);
  conv(Wf1, wf1w, 6 * 256 * 1024);
  conv(Wf2, wf2w, 6 * 1024 * 256);

  // ---- embedding ----
  gather_k<<<(Mn * 256 + 255) / 256, 256, 0, stream>>>(opcode, W_op, hcat);
  gemm_k<true, false, false><<<dim3(2, 64), 256, 0, stream>>>(
      node_feat, wfeat, b_feat, nullptr, nullptr, hcat + 256, Mn, 140, 256, 512);
  gemm_k<false, true, false><<<dim3(2, 64), 256, 0, stream>>>(
      hcat, wmlp1, b_mlp1, nullptr, nullptr, t1, Mn, 512, 256, 256);
  gemm_k<false, false, false><<<dim3(2, 64), 256, 0, stream>>>(
      t1, wmlp2, b_mlp2, nullptr, hbuf, nullptr, Mn, 256, 256, 256);
  assemble_k<<<Mt, 256, 0, stream>>>(hbuf, cls, x, xb);

  // ---- transformer layers ----
  for (int i = 0; i < 6; ++i) {
    const bh16* Wqi = wq + (size_t)i * 256 * 256;
    const bh16* Wki = wk + (size_t)i * 256 * 256;
    const bh16* Wvi = wv + (size_t)i * 256 * 256;
    const bh16* Woi = wo + (size_t)i * 256 * 256;
    gemm_k<false, false, false><<<dim3(2, 64), 256, 0, stream>>>(
        xb, Wqi, bq + i * 256, nullptr, nullptr, qb, Mt, 256, 256, 256);
    gemm_k<false, false, false><<<dim3(2, 64), 256, 0, stream>>>(
        xb, Wki, bk + i * 256, nullptr, nullptr, kb, Mt, 256, 256, 256);
    gemm_k<false, false, false><<<dim3(2, 64), 256, 0, stream>>>(
        xb, Wvi, bv + i * 256, nullptr, nullptr, vb, Mt, 256, 256, 256);
    attn_k<<<dim3(64, 64), 32, 0, stream>>>(qb, kb, vb, dist, dtab, ctxb);
    gemm_k<false, false, true><<<dim3(2, 64), 256, 0, stream>>>(
        ctxb, Woi, bo + i * 256, x, y, nullptr, Mt, 256, 256, 256);
    ln_k<<<Mt / 8, 256, 0, stream>>>(y, ln1g + i * 256, ln1b + i * 256, x, xb);
    gemm_k<false, true, false><<<dim3(8, 64), 256, 0, stream>>>(
        xb, wf1w + (size_t)i * 256 * 1024, bf1 + i * 1024, nullptr, nullptr, ffn,
        Mt, 256, 1024, 1024);
    gemm_k<false, false, true><<<dim3(2, 64), 256, 0, stream>>>(
        ffn, wf2w + (size_t)i * 1024 * 256, bf2 + i * 256, x, y, nullptr,
        Mt, 1024, 256, 256);
    float* xout = (i == 5) ? (float*)d_out : x;   // final LN writes the output
    ln_k<<<Mt / 8, 256, 0, stream>>>(y, ln2g + i * 256, ln2b + i * 256, xout, xb);
  }
}
